// GoldenMoEBaseline_9981503995947
// MI455X (gfx1250) — matrix operand order repaired and checked by reading, not compiled, not run on hardware
//
#include <hip/hip_runtime.h>

#define N_TOKENS 8192
#define D_IN     1024
#define D_HID    4096
#define D_OUT    1024
#define N_EXP    8
#define N_ACT    5

typedef __attribute__((ext_vector_type(16))) __bf16 v16bf;
typedef __attribute__((ext_vector_type(8)))  float  v8f;

union FragAB {
    v16bf        v;
    unsigned int u[8];
    uint4        q[2];
};

__device__ __forceinline__ unsigned short f32_to_bf16_rne(float f) {
    unsigned int u = __float_as_uint(f);
    u += 0x7fffu + ((u >> 16) & 1u);   // round-to-nearest-even
    return (unsigned short)(u >> 16);
}

// CDNA5 async copy: global -> LDS, 16B per lane, tracked by ASYNCcnt.
__device__ __forceinline__ void async_copy_b128(unsigned lds_off, const void* gptr) {
    asm volatile("global_load_async_to_lds_b128 %0, %1, off"
                 :: "v"(lds_off), "v"((unsigned long long)(uintptr_t)gptr)
                 : "memory");
}
__device__ __forceinline__ void wait_async0() {
    asm volatile("s_wait_asynccnt 0x0" ::: "memory");
}

// ---------------------------------------------------------------------------
// x (f32) -> xb (bf16), 4 elements / thread
// ---------------------------------------------------------------------------
__global__ __launch_bounds__(256) void cvt_bf16_kernel(
    const float* __restrict__ src, unsigned short* __restrict__ dst, int n4) {
    int i = blockIdx.x * blockDim.x + threadIdx.x;
    if (i >= n4) return;
    float4 v = ((const float4*)src)[i];
    unsigned int lo = (unsigned int)f32_to_bf16_rne(v.x) |
                      ((unsigned int)f32_to_bf16_rne(v.y) << 16);
    unsigned int hi = (unsigned int)f32_to_bf16_rne(v.z) |
                      ((unsigned int)f32_to_bf16_rne(v.w) << 16);
    uint2 o; o.x = lo; o.y = hi;
    ((uint2*)dst)[i] = o;
}

// ---------------------------------------------------------------------------
// W [K][N] f32 -> Wb [N][K] bf16 (transpose via LDS, vectorized stores).
// Run once per expert; removes f32 reads + conversion from the GEMM hot loop.
// ---------------------------------------------------------------------------
__global__ __launch_bounds__(256) void cvt_w_kernel(
    const float* __restrict__ W, unsigned short* __restrict__ Wb, int K, int N) {
    __shared__ unsigned short tile[64][72];   // padded
    const int n0 = blockIdx.x * 64;
    const int k0 = blockIdx.y * 64;
    const int t = threadIdx.x;
#pragma unroll
    for (int p = 0; p < 16; ++p) {
        int idx = p * 256 + t;          // 0..4095
        int kk = idx >> 6;
        int nn = idx & 63;
        tile[nn][kk] = f32_to_bf16_rne(W[(size_t)(k0 + kk) * N + n0 + nn]);
    }
    __syncthreads();
#pragma unroll
    for (int p = 0; p < 2; ++p) {
        int idx = p * 256 + t;          // 0..511, 16B each
        int nn = idx >> 3;
        int k8 = (idx & 7) * 8;
        uint4 v = *(const uint4*)&tile[nn][k8];
        *(uint4*)(Wb + (size_t)(n0 + nn) * K + k0 + k8) = v;
    }
}

// ---------------------------------------------------------------------------
// Gating: one wave per token. scores = (x@Wg + bg)/e -> softmax -> top-5 mask
// -> renormalize. Writes gw[token][8].
// ---------------------------------------------------------------------------
__global__ __launch_bounds__(256) void gate_kernel(
    const float* __restrict__ x, const float* __restrict__ Wg,
    const float* __restrict__ bg, float* __restrict__ gw) {
    const int lane = threadIdx.x & 31;
    const int wave = threadIdx.x >> 5;
    const int n = blockIdx.x * 8 + wave;
    const float* xr = x + (size_t)n * D_IN;

    float acc[N_EXP];
#pragma unroll
    for (int e = 0; e < N_EXP; ++e) acc[e] = 0.f;
    for (int d = lane; d < D_IN; d += 32) {
        float xv = xr[d];
        const float* wr = Wg + (size_t)d * N_EXP;
#pragma unroll
        for (int e = 0; e < N_EXP; ++e) acc[e] = fmaf(xv, wr[e], acc[e]);
    }
#pragma unroll
    for (int off = 16; off > 0; off >>= 1) {
#pragma unroll
        for (int e = 0; e < N_EXP; ++e) acc[e] += __shfl_xor(acc[e], off, 32);
    }
    const float invT = 0.36787944117144233f;  // 1/e
    float p[N_EXP];
    float mx = -1e30f;
#pragma unroll
    for (int e = 0; e < N_EXP; ++e) {
        p[e] = (acc[e] + bg[e]) * invT;
        mx = fmaxf(mx, p[e]);
    }
    float sum = 0.f;
#pragma unroll
    for (int e = 0; e < N_EXP; ++e) { p[e] = __expf(p[e] - mx); sum += p[e]; }
    float inv = 1.f / sum;
#pragma unroll
    for (int e = 0; e < N_EXP; ++e) p[e] *= inv;

    unsigned sel = 0;
    for (int it = 0; it < N_ACT; ++it) {
        int best = 0; float bv = -1.f;
#pragma unroll
        for (int e = 0; e < N_EXP; ++e)
            if (!((sel >> e) & 1) && p[e] > bv) { bv = p[e]; best = e; }
        sel |= 1u << best;
    }
    float wsum = 0.f;
#pragma unroll
    for (int e = 0; e < N_EXP; ++e) if ((sel >> e) & 1) wsum += p[e];
    float winv = 1.f / (wsum + 1e-8f);
    if (lane < N_EXP)
        gw[(size_t)n * N_EXP + lane] = ((sel >> lane) & 1) ? p[lane] * winv : 0.f;
}

// ---------------------------------------------------------------------------
// Fused expert GEMM. 256 threads = 8 wave32, 128x128 f32 accumulator tile,
// bf16 WMMA 16x16x32. Both operands staged with global_load_async_to_lds_b128
// into double-buffered LDS; next k-chunk in flight during current WMMAs.
//   FUSE_OUT = false : H = relu(A@B + bias)        (bf16 -> hb)
//   FUSE_OUT = true  : Y (+)= gw[m,e]*(A@B + bias) (f32 -> d_out)
// A: [M][KTOT] bf16. Bw: [NTOT][KTOT] bf16 (pre-transposed by cvt_w_kernel).
// ---------------------------------------------------------------------------
template <int KTOT, int NTOT, bool FUSE_OUT>
__global__ __launch_bounds__(256) void moe_gemm_kernel(
    const unsigned short* __restrict__ A, const unsigned short* __restrict__ Bw,
    const float* __restrict__ bias, unsigned short* __restrict__ H,
    const float* __restrict__ gw, float* __restrict__ Y, int expert) {
    __shared__ unsigned short Alds[2][128][40];   // [buf][m][k], padded
    __shared__ unsigned short Blds[2][128][40];   // [buf][n][k], padded

    const int n0 = blockIdx.x * 128;
    const int m0 = blockIdx.y * 128;
    const int t = threadIdx.x;
    const int lane = t & 31;
    const int wave = t >> 5;
    const int wm = wave >> 1;   // 0..3 : 32-row slab
    const int wn = wave & 1;    // 0..1 : 64-col slab
    const int half = lane >> 4;
    const int l16 = lane & 15;

    v8f acc[2][4];
    const v8f vzero = {0.f, 0.f, 0.f, 0.f, 0.f, 0.f, 0.f, 0.f};
#pragma unroll
    for (int i = 0; i < 2; ++i)
#pragma unroll
        for (int j = 0; j < 4; ++j) acc[i][j] = vzero;

    // Staging: 2 threads per row/col, 32B (= 2 x b128 async) each.
    const int sr = t >> 1;
    const int seg = t & 1;
    const unsigned short* agsrc = A  + (size_t)(m0 + sr) * KTOT + seg * 16;
    const unsigned short* bgsrc = Bw + (size_t)(n0 + sr) * KTOT + seg * 16;

    auto stage = [&](int buf, int k0) {
        unsigned aoff = (unsigned)(uintptr_t)&Alds[buf][sr][seg * 16];
        unsigned boff = (unsigned)(uintptr_t)&Blds[buf][sr][seg * 16];
        const unsigned short* ag = agsrc + k0;
        const unsigned short* bg = bgsrc + k0;
        async_copy_b128(aoff,      ag);
        async_copy_b128(aoff + 16, ag + 8);
        async_copy_b128(boff,      bg);
        async_copy_b128(boff + 16, bg + 8);
    };

    stage(0, 0);
    wait_async0();
    __syncthreads();

    for (int k0 = 0; k0 < KTOT; k0 += 32) {
        const int cur = (k0 >> 5) & 1;
        const bool more = (k0 + 32 < KTOT);
        if (more) stage(cur ^ 1, k0 + 32);   // overlap with compute below

        // --- fragments: contiguous k = 16*half + s in both A and B ---
        FragAB a[2], b[4];
#pragma unroll
        for (int i = 0; i < 2; ++i) {
            int row = wm * 32 + i * 16 + l16;
#pragma unroll
            for (int j = 0; j < 8; ++j) {
                // lanes 0-15: K 0..15; lanes 16-31: K 16..31
                int kb = 2 * j + 16 * half;
                a[i].u[j] = *(const unsigned int*)&Alds[cur][row][kb];
            }
        }
#pragma unroll
        for (int j = 0; j < 4; ++j) {
            int col = wn * 64 + j * 16 + l16;
            const uint4* bp = (const uint4*)&Blds[cur][col][16 * half];
            b[j].q[0] = bp[0];
            b[j].q[1] = bp[1];
        }
#pragma unroll
        for (int i = 0; i < 2; ++i)
#pragma unroll
            for (int j = 0; j < 4; ++j)
                acc[i][j] = __builtin_amdgcn_wmma_f32_16x16x32_bf16(
                    false, a[i].v, false, b[j].v, (short)0, acc[i][j], false, false);

        if (more) {
            wait_async0();       // next buffer fully in LDS (this wave)
            __syncthreads();     // ... and for all waves; cur free for re-stage
        }
    }

    // --- epilogue ---
#pragma unroll
    for (int i = 0; i < 2; ++i) {
#pragma unroll
        for (int j = 0; j < 4; ++j) {
            const int n = n0 + wn * 64 + j * 16 + l16;
            const float bv = bias[n];
#pragma unroll
            for (int r = 0; r < 8; ++r) {
                const int m = m0 + wm * 32 + i * 16 + half * 8 + r;
                float v = acc[i][j][r] + bv;
                if (!FUSE_OUT) {
                    v = v > 0.f ? v : 0.f;  // ReLU
                    H[(size_t)m * NTOT + n] = f32_to_bf16_rne(v);
                } else {
                    const float w = gw[(size_t)m * N_EXP + expert];
                    const size_t idx = (size_t)m * NTOT + n;
                    const float val = v * w;
                    if (expert == 0) Y[idx] = val;   // initializes d_out
                    else             Y[idx] += val;  // experts run sequentially
                }
            }
        }
    }
}

// ---------------------------------------------------------------------------
extern "C" void kernel_launch(void* const* d_in, const int* in_sizes, int n_in,
                              void* d_out, int out_size, void* d_ws, size_t ws_size,
                              hipStream_t stream) {
    (void)in_sizes; (void)n_in; (void)out_size; (void)ws_size;
    const float* x  = (const float*)d_in[0];
    const float* W1 = (const float*)d_in[1];
    const float* b1 = (const float*)d_in[2];
    const float* W2 = (const float*)d_in[3];
    const float* b2 = (const float*)d_in[4];
    const float* Wg = (const float*)d_in[5];
    const float* bg = (const float*)d_in[6];
    float* y = (float*)d_out;

    char* ws = (char*)d_ws;
    size_t off = 0;
    float* gate_w = (float*)(ws + off);          off += (size_t)N_TOKENS * N_EXP * 4;   // 256 KB
    unsigned short* xb  = (unsigned short*)(ws + off); off += (size_t)N_TOKENS * D_IN  * 2; // 16.8 MB
    unsigned short* hb  = (unsigned short*)(ws + off); off += (size_t)N_TOKENS * D_HID * 2; // 67 MB
    unsigned short* wb1 = (unsigned short*)(ws + off); off += (size_t)D_IN  * D_HID * 2;    // 8.4 MB
    unsigned short* wb2 = (unsigned short*)(ws + off);                                      // 8.4 MB

    cvt_bf16_kernel<<<(N_TOKENS * D_IN / 4 + 255) / 256, 256, 0, stream>>>(
        x, xb, N_TOKENS * D_IN / 4);
    gate_kernel<<<N_TOKENS / 8, 256, 0, stream>>>(x, Wg, bg, gate_w);

    for (int e = 0; e < N_EXP; ++e) {
        cvt_w_kernel<<<dim3(D_HID / 64, D_IN / 64), 256, 0, stream>>>(
            W1 + (size_t)e * D_IN * D_HID, wb1, D_IN, D_HID);
        cvt_w_kernel<<<dim3(D_OUT / 64, D_HID / 64), 256, 0, stream>>>(
            W2 + (size_t)e * D_HID * D_OUT, wb2, D_HID, D_OUT);
        moe_gemm_kernel<D_IN, D_HID, false>
            <<<dim3(D_HID / 128, N_TOKENS / 128), 256, 0, stream>>>(
                xb, wb1, b1 + (size_t)e * D_HID, hb, nullptr, nullptr, e);
        moe_gemm_kernel<D_HID, D_OUT, true>
            <<<dim3(D_OUT / 128, N_TOKENS / 128), 256, 0, stream>>>(
                hb, wb2, b2 + (size_t)e * D_OUT, nullptr, gate_w, y, e);
    }
}
